// DNNTSP_67929202753598
// MI455X (gfx1250) — compile-verified, matrix operand order checked by reading, not run on hardware
//
#include <hip/hip_runtime.h>
#include <hip/hip_bf16.h>

#define ITEMS 1024
#define DD    128
#define HEADS 4
#define BBAT  16
#define NN    (BBAT * ITEMS)     // 16384 nodes
#define QD    (HEADS * DD)       // 512
#define BNEPS 1e-5f

typedef __attribute__((ext_vector_type(16))) __bf16 v16bf;
typedef __attribute__((ext_vector_type(8)))  float  v8f;

__device__ __forceinline__ __bf16 f2bf(float f) {
    union { float f; unsigned u; } x; x.f = f;
    unsigned r = (x.u + 0x7FFFu + ((x.u >> 16) & 1u)) >> 16;
    union { unsigned short s; __bf16 b; } y; y.s = (unsigned short)r;
    return y.b;
}

// Load a 16x32 A-fragment (or matching B-fragment for C = X * W^T, since
// B column n == W row n) from a row-major bf16 matrix. ISA 7.12.2 layout:
// lane l holds row (l&15); element e holds K = (e>>3)*16 + ((l>=16)?8:0) + (e&7).
__device__ __forceinline__ v16bf load_frag16(const __bf16* rowPtr, int kBase, int lane) {
    const int khalf = (lane & 16) >> 1;  // 8 for lanes 16..31
    v16bf f;
#pragma unroll
    for (int e = 0; e < 16; ++e) {
        int k = ((e >> 3) << 4) + khalf + (e & 7);
        f[e] = rowPtr[kBase + k];
    }
    return f;
}

__device__ __forceinline__ v8f wmma_bf16(v16bf a, v16bf b, v8f c) {
    return __builtin_amdgcn_wmma_f32_16x16x32_bf16(false, a, false, b, (short)0, c,
                                                   false, false);
}

// ---------------------------------------------------------------- elementwise
__global__ void cvt_bf16_kernel(const float* __restrict__ src, __bf16* __restrict__ dst, int n) {
    int i = blockIdx.x * 256 + threadIdx.x;
    if (i < n) dst[i] = f2bf(src[i]);
}

__global__ void fill_f32_kernel(float* __restrict__ p, float v, int n) {
    int i = blockIdx.x * 256 + threadIdx.x;
    if (i < n) p[i] = v;
}

__global__ void deg_accum_kernel(const int* __restrict__ col, const float* __restrict__ ew,
                                 float* __restrict__ deg, int E) {
    int i = blockIdx.x * 256 + threadIdx.x;
    if (i < E) atomicAdd(&deg[col[i]], ew[i]);
}

__global__ void dis_kernel(float* __restrict__ deg, int n) {
    int i = blockIdx.x * 256 + threadIdx.x;
    if (i < n) {
        float d = deg[i];
        deg[i] = (d > 0.0f) ? rsqrtf(fmaxf(d, 1e-12f)) : 0.0f;
    }
}

// out = self-loop term + bias:  out[i,d] = dis[i]^2 * xw[i,d] + b[d]
__global__ void scatter_init_kernel(float* __restrict__ out, const float* __restrict__ xw,
                                    const float* __restrict__ dis, const float* __restrict__ bias) {
    int i = blockIdx.x * 256 + threadIdx.x;  // N*D threads
    int node = i >> 7, d = i & 127;
    float s = dis[node];
    out[i] = s * s * xw[i] + bias[d];
}

// out[c] += dis[r]*ew*dis[c] * xw[r]   (2 edges per 256-thread block)
__global__ void scatter_edge_kernel(const int* __restrict__ row, const int* __restrict__ col,
                                    const float* __restrict__ ew, const float* __restrict__ dis,
                                    const float* __restrict__ xw, float* __restrict__ out, int E) {
    int e = (blockIdx.x << 1) + (threadIdx.x >> 7);
    int d = threadIdx.x & 127;
    if (e >= E) return;
    int r = row[e], c = col[e];
    const float* src = xw + (size_t)r * DD;
    __builtin_prefetch(src, 0, 0);  // global_prefetch
    float w = dis[r] * ew[e] * dis[c];
    atomicAdd(&out[(size_t)c * DD + d], w * src[d]);
}

__global__ void bn_stats_kernel(const float* __restrict__ h, float* __restrict__ sum,
                                float* __restrict__ sumsq) {
    int d  = threadIdx.x;            // 128 columns
    int r0 = blockIdx.x * 128;       // 128 rows per block
    float s = 0.0f, q = 0.0f;
    for (int r = 0; r < 128; ++r) {
        float v = h[(size_t)(r0 + r) * DD + d];
        s += v; q += v * v;
    }
    atomicAdd(&sum[d], s);
    atomicAdd(&sumsq[d], q);
}

__global__ void bn_relu_kernel(const float* __restrict__ h, const float* __restrict__ sum,
                               const float* __restrict__ sumsq, const float* __restrict__ g,
                               const float* __restrict__ bb, __bf16* __restrict__ hb) {
    int i = blockIdx.x * 256 + threadIdx.x;  // N*D
    int d = i & 127;
    const float invN = 1.0f / (float)NN;
    float mu  = sum[d] * invN;
    float var = sumsq[d] * invN - mu * mu;   // biased variance, matches jnp.var
    float rs  = rsqrtf(var + BNEPS);
    float v   = (h[i] - mu) * rs * g[d] + bb[d];
    hb[i] = f2bf(fmaxf(v, 0.0f));
}

__global__ void final_gate_kernel(const float* __restrict__ agg, const float* __restrict__ emb,
                                  const float* __restrict__ alpha, float* __restrict__ out) {
    int i = blockIdx.x * 256 + threadIdx.x;  // B*ITEMS*D
    int item = (i >> 7) & (ITEMS - 1);
    int d = i & 127;
    float a = alpha[item];
    out[i] = (1.0f - a) * emb[(size_t)item * DD + d] + a * agg[i];
}

// ------------------------------------------------------------------ WMMA GEMM
// C[M,Nout] = A[M,128] * W[Nout,128]^T ; one wave per 16x16 tile, K=128.
// Templated on output type to keep the store epilogue branch-free.
template <bool BF16OUT>
__global__ void gemm_bf16_kernel(const __bf16* __restrict__ A, const __bf16* __restrict__ W,
                                 void* __restrict__ outP, int M, int Nout) {
    int wave = (blockIdx.x << 3) + (threadIdx.x >> 5);  // 8 waves / block
    int lane = threadIdx.x & 31;
    int ntile = Nout >> 4;
    int tiles = (M >> 4) * ntile;
    if (wave >= tiles) return;                            // wave-uniform
    int tm = wave / ntile, tn = wave - tm * ntile;
    const __bf16* aRow = A + (size_t)(tm * 16 + (lane & 15)) * DD;
    const __bf16* wRow = W + (size_t)(tn * 16 + (lane & 15)) * DD;
    v8f c = {};
#pragma unroll
    for (int kb = 0; kb < DD; kb += 32) {
        v16bf a = load_frag16(aRow, kb, lane);
        v16bf b = load_frag16(wRow, kb, lane);
        c = wmma_bf16(a, b, c);
    }
    int half8 = (lane & 16) >> 1;   // row offset 8 for lanes 16..31
    int col = tn * 16 + (lane & 15);
#pragma unroll
    for (int v = 0; v < 8; ++v) {
        size_t idx = (size_t)(tm * 16 + v + half8) * Nout + col;
        if (BF16OUT) ((__bf16*)outP)[idx] = f2bf(c[v]);
        else         ((float*)outP)[idx]  = c[v];
    }
}

// ------------------------------------------------------- flash causal attention
// One wave per (batch b, head h, 16-row query tile). Online softmax over
// 32-key superblocks; scores and PV both via v_wmma_f32_16x16x32_bf16.
// The 32x128 V tile is staged into LDS with async DMA (global_load_async_to_lds)
// issued before the score WMMAs so the copy overlaps the matrix math.
// Pinned at 1 wave/SIMD so the ~112 VGPRs of persistent state (O accum, Q frags,
// softmax stats) stay resident -- the 4096 independent blocks fill the WGPs.
__global__ __launch_bounds__(32, 1)
__attribute__((amdgpu_waves_per_eu(1)))
void flash_attn_kernel(const __bf16* __restrict__ qB, const __bf16* __restrict__ kB,
                       const __bf16* __restrict__ vB, float* __restrict__ attnF) {
    __shared__ float  pTile[16 * 32];      // P scores, C-layout staging
    __shared__ __bf16 vTile[32 * 128];     // V superblock, row-major [k][d]
    int lane = threadIdx.x & 31;
    int id = blockIdx.x;
    int qT = id & 63;
    int hh = (id >> 6) & 3;
    int b  = id >> 8;
    int khalf = (lane & 16) >> 1;  // 8 for high half
    int colL  = lane & 15;

    const size_t seqBase = (size_t)b * ITEMS;
    const __bf16* qRow = qB + (seqBase + qT * 16 + colL) * QD + hh * DD;

    // LDS byte offset of vTile (flat LDS addresses carry the offset in the low bits)
    unsigned vLds = (unsigned)(unsigned long long)(&vTile[0]) + (unsigned)(lane * 256);

    v16bf qf[4];
#pragma unroll
    for (int kk = 0; kk < 4; ++kk) qf[kk] = load_frag16(qRow, kk * 32, lane);

    v8f z = {};
    v8f o[8];
#pragma unroll
    for (int dc = 0; dc < 8; ++dc) o[dc] = z;
    float mrow[8], lrow[8];
#pragma unroll
    for (int v = 0; v < 8; ++v) { mrow[v] = -1e30f; lrow[v] = 0.0f; }

    const float scale = 0.08838834764831845f;  // 1/sqrt(128)
    int nSB = (qT + 2) >> 1;
    for (int sb = 0; sb < nSB; ++sb) {
        int t0 = sb * 32;

        // --- async-stage V[t0..t0+31][hh*128..+128] into LDS (coalesced 16B chunks,
        //     one row per lane); overlaps with the score WMMAs below.
        {
            const __bf16* gv = vB + (seqBase + t0 + lane) * QD + hh * DD;
#pragma unroll
            for (int ch = 0; ch < 16; ++ch) {
                unsigned dst = vLds + (unsigned)(ch * 16);
                const void* src = (const void*)(gv + ch * 8);
                asm volatile("global_load_async_to_lds_b128 %0, %1, off"
                             :: "v"(dst), "v"(src) : "memory");
            }
        }

        // --- S = Q K^T for two 16-key subtiles
        const __bf16* kRow0 = kB + (seqBase + t0 + colL) * QD + hh * DD;
        const __bf16* kRow1 = kRow0 + (size_t)16 * QD;
        v8f s0 = z, s1 = z;
#pragma unroll
        for (int kk = 0; kk < 4; ++kk)
            s0 = wmma_bf16(qf[kk], load_frag16(kRow0, kk * 32, lane), s0);
#pragma unroll
        for (int kk = 0; kk < 4; ++kk)
            s1 = wmma_bf16(qf[kk], load_frag16(kRow1, kk * 32, lane), s1);

        // scale + causal mask (C layout: row = v + khalf, col = colL)
#pragma unroll
        for (int v = 0; v < 8; ++v) {
            int qg = qT * 16 + v + khalf;
            float a0 = s0[v] * scale, a1 = s1[v] * scale;
            if (t0 + colL > qg)      a0 = -1e30f;
            if (t0 + 16 + colL > qg) a1 = -1e30f;
            s0[v] = a0; s1[v] = a1;
        }

        // online softmax: per-row reductions live in 16-lane groups
#pragma unroll
        for (int v = 0; v < 8; ++v) {
            float mloc = fmaxf(s0[v], s1[v]);
#pragma unroll
            for (int off = 1; off < 16; off <<= 1)
                mloc = fmaxf(mloc, __shfl_xor(mloc, off, 32));
            float mnew = fmaxf(mrow[v], mloc);
            float corr = __expf(mrow[v] - mnew);
            float p0 = __expf(s0[v] - mnew);
            float p1 = __expf(s1[v] - mnew);
            float rs = p0 + p1;
#pragma unroll
            for (int off = 1; off < 16; off <<= 1)
                rs += __shfl_xor(rs, off, 32);
            lrow[v] = lrow[v] * corr + rs;
            mrow[v] = mnew;
#pragma unroll
            for (int dc = 0; dc < 8; ++dc) o[dc][v] *= corr;
            int m = v + khalf;
            pTile[m * 32 + colL]      = p0;   // C-layout -> LDS 16x32 tile
            pTile[m * 32 + colL + 16] = p1;
        }
        // make async V staging + P tile visible before LDS reads
        asm volatile("s_wait_asynccnt 0x0" ::: "memory");
        __syncthreads();

        // re-read P in A-fragment layout (bf16)
        v16bf pa;
#pragma unroll
        for (int e = 0; e < 16; ++e) {
            int k = ((e >> 3) << 4) + khalf + (e & 7);
            pa[e] = f2bf(pTile[colL * 32 + k]);
        }

        // O += P(16x32) * V(32x16) per 16-wide D chunk, V read from LDS
#pragma unroll
        for (int dc = 0; dc < 8; ++dc) {
            v16bf bf;
#pragma unroll
            for (int e = 0; e < 16; ++e) {
                int k = ((e >> 3) << 4) + khalf + (e & 7);
                bf[e] = vTile[k * DD + dc * 16 + colL];
            }
            o[dc] = wmma_bf16(pa, bf, o[dc]);
        }
        __syncthreads();
    }

    // normalize and accumulate head mean (x0.25) into attnF
#pragma unroll
    for (int dc = 0; dc < 8; ++dc) {
#pragma unroll
        for (int v = 0; v < 8; ++v) {
            int m = v + khalf;
            size_t grow = seqBase + qT * 16 + m;
            atomicAdd(attnF + grow * DD + dc * 16 + colL, (o[dc][v] / lrow[v]) * 0.25f);
        }
    }
}

// ------------------------------------------------------------------- launcher
static inline size_t alignUp256(size_t x) { return (x + 255) & ~(size_t)255; }

extern "C" void kernel_launch(void* const* d_in, const int* in_sizes, int n_in,
                              void* d_out, int out_size, void* d_ws, size_t ws_size,
                              hipStream_t stream) {
    const float* X     = (const float*)d_in[0];
    const int*   ei    = (const int*)d_in[1];
    const float* ew    = (const float*)d_in[2];
    const float* emb   = (const float*)d_in[3];
    const float* W1    = (const float*)d_in[4];
    const float* b1    = (const float*)d_in[5];
    const float* g1    = (const float*)d_in[6];
    const float* bb1   = (const float*)d_in[7];
    const float* W2    = (const float*)d_in[8];
    const float* b2    = (const float*)d_in[9];
    const float* g2    = (const float*)d_in[10];
    const float* bb2   = (const float*)d_in[11];
    const float* Wq    = (const float*)d_in[12];
    const float* Wk    = (const float*)d_in[13];
    const float* Wv    = (const float*)d_in[14];
    const float* aggW  = (const float*)d_in[15];
    const float* alpha = (const float*)d_in[16];
    const int E = in_sizes[2];
    const int* rowI = ei;
    const int* colI = ei + E;
    float* out = (float*)d_out;

    char* w = (char*)d_ws;
    auto take = [&](size_t bytes) -> void* { void* p = (void*)w; w += alignUp256(bytes); return p; };

    __bf16* Xb    = (__bf16*)take((size_t)NN * DD * 2);
    __bf16* W1b   = (__bf16*)take((size_t)DD * DD * 2);
    __bf16* W2b   = (__bf16*)take((size_t)DD * DD * 2);
    __bf16* Wqb   = (__bf16*)take((size_t)QD * DD * 2);
    __bf16* Wkb   = (__bf16*)take((size_t)QD * DD * 2);
    __bf16* Wvb   = (__bf16*)take((size_t)QD * DD * 2);
    __bf16* aggWb = (__bf16*)take((size_t)DD * DD * 2);
    float*  deg   = (float*)take((size_t)NN * 4);           // becomes dis in place
    float*  xw    = (float*)take((size_t)NN * DD * 4);
    float*  hbuf  = (float*)take((size_t)NN * DD * 4);
    float*  sumv  = (float*)take((size_t)DD * 4);
    float*  sumq  = (float*)take((size_t)DD * 4);
    __bf16* hb    = (__bf16*)take((size_t)NN * DD * 2);
    __bf16* qb    = (__bf16*)take((size_t)NN * QD * 2);
    __bf16* kb    = (__bf16*)take((size_t)NN * QD * 2);
    __bf16* vb    = (__bf16*)take((size_t)NN * QD * 2);
    float*  attnF = (float*)take((size_t)NN * DD * 4);
    __bf16* attnB = (__bf16*)take((size_t)NN * DD * 2);
    float*  aggF  = (float*)take((size_t)NN * DD * 4);
    (void)ws_size; (void)n_in; (void)out_size;

    const int nd = NN * DD;
    auto g256 = [](int n) { return (n + 255) / 256; };

    // 1) bf16 conversions of activations + weights
    cvt_bf16_kernel<<<g256(nd), 256, 0, stream>>>(X, Xb, nd);
    cvt_bf16_kernel<<<g256(DD * DD), 256, 0, stream>>>(W1, W1b, DD * DD);
    cvt_bf16_kernel<<<g256(DD * DD), 256, 0, stream>>>(W2, W2b, DD * DD);
    cvt_bf16_kernel<<<g256(QD * DD), 256, 0, stream>>>(Wq, Wqb, QD * DD);
    cvt_bf16_kernel<<<g256(QD * DD), 256, 0, stream>>>(Wk, Wkb, QD * DD);
    cvt_bf16_kernel<<<g256(QD * DD), 256, 0, stream>>>(Wv, Wvb, QD * DD);
    cvt_bf16_kernel<<<g256(DD * DD), 256, 0, stream>>>(aggW, aggWb, DD * DD);

    // 2) degrees (self loop contributes 1) -> dis = rsqrt(deg)
    fill_f32_kernel<<<g256(NN), 256, 0, stream>>>(deg, 1.0f, NN);
    deg_accum_kernel<<<g256(E), 256, 0, stream>>>(colI, ew, deg, E);
    dis_kernel<<<g256(NN), 256, 0, stream>>>(deg, NN);

    const int tilesD  = (NN / 16) * (DD / 16);
    const int tilesQD = (NN / 16) * (QD / 16);

    // 3) GCN layer 1
    gemm_bf16_kernel<false><<<(tilesD + 7) / 8, 256, 0, stream>>>(Xb, W1b, xw, NN, DD);
    scatter_init_kernel<<<g256(nd), 256, 0, stream>>>(hbuf, xw, deg, b1);
    scatter_edge_kernel<<<(E + 1) / 2, 256, 0, stream>>>(rowI, colI, ew, deg, xw, hbuf, E);
    fill_f32_kernel<<<1, 256, 0, stream>>>(sumv, 0.0f, DD);
    fill_f32_kernel<<<1, 256, 0, stream>>>(sumq, 0.0f, DD);
    bn_stats_kernel<<<NN / 128, 128, 0, stream>>>(hbuf, sumv, sumq);
    bn_relu_kernel<<<g256(nd), 256, 0, stream>>>(hbuf, sumv, sumq, g1, bb1, hb);

    // 4) GCN layer 2
    gemm_bf16_kernel<false><<<(tilesD + 7) / 8, 256, 0, stream>>>(hb, W2b, xw, NN, DD);
    scatter_init_kernel<<<g256(nd), 256, 0, stream>>>(hbuf, xw, deg, b2);
    scatter_edge_kernel<<<(E + 1) / 2, 256, 0, stream>>>(rowI, colI, ew, deg, xw, hbuf, E);
    fill_f32_kernel<<<1, 256, 0, stream>>>(sumv, 0.0f, DD);
    fill_f32_kernel<<<1, 256, 0, stream>>>(sumq, 0.0f, DD);
    bn_stats_kernel<<<NN / 128, 128, 0, stream>>>(hbuf, sumv, sumq);
    bn_relu_kernel<<<g256(nd), 256, 0, stream>>>(hbuf, sumv, sumq, g2, bb2, hb);

    // 5) QKV projections (bf16 outputs for the attention WMMAs)
    gemm_bf16_kernel<true><<<(tilesQD + 7) / 8, 256, 0, stream>>>(hb, Wqb, qb, NN, QD);
    gemm_bf16_kernel<true><<<(tilesQD + 7) / 8, 256, 0, stream>>>(hb, Wkb, kb, NN, QD);
    gemm_bf16_kernel<true><<<(tilesQD + 7) / 8, 256, 0, stream>>>(hb, Wvb, vb, NN, QD);

    // 6) flash causal attention, head-mean accumulated into attnF
    fill_f32_kernel<<<g256(nd), 256, 0, stream>>>(attnF, 0.0f, nd);
    flash_attn_kernel<<<BBAT * HEADS * (ITEMS / 16), 32, 0, stream>>>(qb, kb, vb, attnF);

    // 7) aggregate projection + gated update
    cvt_bf16_kernel<<<g256(nd), 256, 0, stream>>>(attnF, attnB, nd);
    gemm_bf16_kernel<false><<<(tilesD + 7) / 8, 256, 0, stream>>>(attnB, aggWb, aggF, NN, DD);
    final_gate_kernel<<<g256(nd), 256, 0, stream>>>(aggF, emb, alpha, out);
}
